// KalmanGainCalculatorMambaSimple_14164802142692
// MI455X (gfx1250) — matrix operand order repaired and checked by reading, not compiled, not run on hardware
//
#include <hip/hip_runtime.h>
#include <hip/hip_bf16.h>
#include <math.h>

#define DIMC    96
#define D_INNER 192
#define D_STATE 16
#define DT_RANK 6
#define KDIR    4
#define BATCH   2
#define HH      64
#define WW      64
#define LL      (HH*WW)
#define NPROJ   38      // DT_RANK + 2*D_STATE
#define NPAD    48      // padded to multiple of 16 for WMMA

typedef __attribute__((ext_vector_type(16))) _Float16 v16h;
typedef __attribute__((ext_vector_type(8)))  float    v8f;

union HV { v16h v; _Float16 h[16]; };
union FV { v8f  v; float    f[8];  };

// ---------------------------------------------------------------------------
// Fragment helpers: A 16x32 f16 packing used here (contiguous k, not AMD's table order) — lane = M%16 (both halves), VGPR v
// holds K pair: kA = (lane>>4)*16 + 2v.
// B 32x16 f16 packing (same k order as A) — lane%16 = N column, kB = (lane>>4)*16 + 2v.
// Fragments stored pre-swizzled: [tile][kstep][lane][16 halves] = 32B/lane.
// ---------------------------------------------------------------------------

// Batched WMMA GEMM on fragment-packed operands. One wave per 16x16 C tile.
// Inner loop: 2 x v16h loads (4 x global_load_b128) + 1 x v_wmma.
__global__ void wmma_gemm_frag(const v16h* __restrict__ A,
                               const v16h* __restrict__ Bm,
                               float* __restrict__ C,
                               int tilesM, int tilesN, int nK, int ldc,
                               long long strideA, long long strideB, long long strideC,
                               int bModB)
{
    const int bidx = blockIdx.y;
    A  += (long long)bidx * strideA;              // strides in v16h chunks
    Bm += (long long)(bidx % bModB) * strideB;
    C  += (long long)bidx * strideC;              // stride in floats

    int wave = (blockIdx.x * blockDim.x + threadIdx.x) >> 5;
    int lane = threadIdx.x & 31;
    if (wave >= tilesM * tilesN) return;          // wave-uniform: EXEC stays all-ones
    int tm = wave / tilesN;
    int tn = wave % tilesN;

    const v16h* Ap = A  + ((long long)tm * nK) * 32 + lane;
    const v16h* Bp = Bm + ((long long)tn * nK) * 32 + lane;

    FV acc;
    #pragma unroll
    for (int i = 0; i < 8; ++i) acc.f[i] = 0.f;

    for (int kt = 0; kt < nK; ++kt) {
        v16h a = Ap[kt * 32];
        v16h b = Bp[kt * 32];
        acc.v = __builtin_amdgcn_wmma_f32_16x16x32_f16(
            false, a, false, b, (short)0, acc.v, false, false);
    }

    int half = lane >> 4, l16 = lane & 15;
    int ncol = tn * 16 + l16;
    #pragma unroll
    for (int v = 0; v < 8; ++v) {
        int m = tm * 16 + v + 8 * half;           // C/D layout: m = vgpr + 8*half
        C[(long long)m * ldc + ncol] = acc.f[v];
    }
}

// ---- pack x (B,96,H,W) -> A fragments for GEMM0 ---------------------------
__global__ void pack_xA_frag(const float* __restrict__ x, v16h* __restrict__ A0)
{
    int t = blockIdx.x * blockDim.x + threadIdx.x;    // (tm*nK + kt)*32 + lane
    const int nK = DIMC / 32;
    const int tilesM = BATCH * LL / 16;
    if (t >= tilesM * nK * 32) return;
    int lane = t & 31;
    int rest = t >> 5;
    int kt = rest % nK, tm = rest / nK;
    int half = lane >> 4, l16 = lane & 15;
    int mrow = tm * 16 + l16;
    int b = mrow / LL, l = mrow % LL;
    HV a;
    #pragma unroll
    for (int v = 0; v < 8; ++v) {
        int c = kt * 32 + half * 16 + 2 * v;
        a.h[2*v]   = (_Float16)x[((long long)b * DIMC + c)     * LL + l];
        a.h[2*v+1] = (_Float16)x[((long long)b * DIMC + c + 1) * LL + l];
    }
    A0[t] = a.v;
}

// ---- pack in_proj_w (384,96) -> B fragments (transposed) ------------------
__global__ void pack_Bin_frag(const float* __restrict__ w, v16h* __restrict__ B0)
{
    int t = blockIdx.x * blockDim.x + threadIdx.x;    // (tn*nK + kt)*32 + lane
    const int nK = DIMC / 32;
    const int tilesN = 2 * D_INNER / 16;
    if (t >= tilesN * nK * 32) return;
    int lane = t & 31;
    int rest = t >> 5;
    int kt = rest % nK, tn = rest / nK;
    int half = lane >> 4, l16 = lane & 15;
    int n = tn * 16 + l16;
    HV bfr;
    #pragma unroll
    for (int v = 0; v < 8; ++v) {
        int k = kt * 32 + half * 16 + 2 * v;
        bfr.h[2*v]   = (_Float16)w[(long long)n * DIMC + k];
        bfr.h[2*v+1] = (_Float16)w[(long long)n * DIMC + k + 1];
    }
    B0[t] = bfr.v;
}

// ---- depthwise 3x3 conv (SAME) + bias + SiLU ------------------------------
__global__ void dwconv_silu(const float* __restrict__ C0,
                            const float* __restrict__ cw,
                            const float* __restrict__ cb,
                            float* __restrict__ xc)
{
    int i = blockIdx.x * blockDim.x + threadIdx.x;   // (b*D_INNER + d)*L + l
    if (i >= BATCH * D_INNER * LL) return;
    int l = i % LL;
    int rest = i / LL;
    int d = rest % D_INNER;
    int b = rest / D_INNER;
    int h = l / WW, w = l % WW;
    float s = cb[d];
    #pragma unroll
    for (int di = -1; di <= 1; ++di)
        #pragma unroll
        for (int dj = -1; dj <= 1; ++dj) {
            int hh = h + di, ww2 = w + dj;
            if (hh >= 0 && hh < HH && ww2 >= 0 && ww2 < WW)
                s += C0[((long long)b * LL + hh * WW + ww2) * (2 * D_INNER) + d]
                     * cw[d * 9 + (di + 1) * 3 + (dj + 1)];
        }
    xc[i] = s / (1.f + __expf(-s));
}

// ---- build 4-direction xs (fp32, d-major, for the scan's u) ---------------
__global__ void build_xs(const float* __restrict__ xc, float* __restrict__ xsf)
{
    int i = blockIdx.x * blockDim.x + threadIdx.x;   // (bk*D_INNER + d)*L + l
    if (i >= BATCH * KDIR * D_INNER * LL) return;
    int l = i % LL;
    int rest = i / LL;
    int d = rest % D_INNER;
    int bk = rest / D_INNER;
    int k = bk % KDIR;
    int b = bk / KDIR;
    int idx = (k >= 2) ? (LL - 1 - l) : l;                       // flip for k=2,3
    int src = (k & 1) ? ((idx % HH) * WW + idx / HH) : idx;      // WH transpose k=1,3
    xsf[i] = xc[((long long)b * D_INNER + d) * LL + src];
}

// ---- pack xs -> A fragments for the x_dbl GEMM (per (b,k) batch) ----------
__global__ void pack_xs_frag(const float* __restrict__ xc, v16h* __restrict__ xsh)
{
    int t = blockIdx.x * blockDim.x + threadIdx.x;   // ((bk*tilesM+tm)*nK+kt)*32+lane
    const int nK = D_INNER / 32;                     // 6
    const int tilesM = LL / 16;                      // 256
    if (t >= BATCH * KDIR * tilesM * nK * 32) return;
    int lane = t & 31;
    int rest = t >> 5;
    int kt = rest % nK;       rest /= nK;
    int tm = rest % tilesM;
    int bk = rest / tilesM;
    int k = bk % KDIR, b = bk / KDIR;
    int half = lane >> 4, l16 = lane & 15;
    int l = tm * 16 + l16;                           // row of per-batch A = seq pos
    int idx = (k >= 2) ? (LL - 1 - l) : l;
    int src = (k & 1) ? ((idx % HH) * WW + idx / HH) : idx;
    const float* base = xc + (long long)b * D_INNER * LL + src;
    HV a;
    #pragma unroll
    for (int v = 0; v < 8; ++v) {
        int d = kt * 32 + half * 16 + 2 * v;
        a.h[2*v]   = (_Float16)base[(long long)d       * LL];
        a.h[2*v+1] = (_Float16)base[(long long)(d + 1) * LL];
    }
    xsh[t] = a.v;
}

// ---- pack x_proj_w (K,38,192) -> B fragments, zero-padded to 48 cols ------
__global__ void pack_xproj_frag(const float* __restrict__ xpw, v16h* __restrict__ Bp)
{
    int t = blockIdx.x * blockDim.x + threadIdx.x;   // ((k*tilesN+tn)*nK+kt)*32+lane
    const int nK = D_INNER / 32;                     // 6
    const int tilesN = NPAD / 16;                    // 3
    if (t >= KDIR * tilesN * nK * 32) return;
    int lane = t & 31;
    int rest = t >> 5;
    int kt = rest % nK;      rest /= nK;
    int tn = rest % tilesN;
    int k  = rest / tilesN;
    int half = lane >> 4, l16 = lane & 15;
    int c = tn * 16 + l16;                           // output column (0..47)
    HV bfr;
    #pragma unroll
    for (int v = 0; v < 8; ++v) {
        int d = kt * 32 + half * 16 + 2 * v;
        float v0 = (c < NPROJ) ? xpw[((long long)k * NPROJ + c) * D_INNER + d]     : 0.f;
        float v1 = (c < NPROJ) ? xpw[((long long)k * NPROJ + c) * D_INNER + d + 1] : 0.f;
        bfr.h[2*v]   = (_Float16)v0;
        bfr.h[2*v+1] = (_Float16)v1;
    }
    Bp[t] = bfr.v;
}

// ---- delta = softplus(dt_projs_w @ dts + dt_projs_b) -----------------------
__global__ void compute_delta(const float* __restrict__ xdbl,
                              const float* __restrict__ dtw,
                              const float* __restrict__ dtb,
                              float* __restrict__ delta)
{
    int i = blockIdx.x * blockDim.x + threadIdx.x;   // (bk*D_INNER + d)*L + l
    if (i >= BATCH * KDIR * D_INNER * LL) return;
    int l = i % LL;
    int rest = i / LL;
    int d = rest % D_INNER;
    int bk = rest / D_INNER;
    int k = bk % KDIR;
    const float* row  = xdbl + ((long long)bk * LL + l) * NPAD;
    const float* wrow = dtw + ((long long)k * D_INNER + d) * DT_RANK;
    float s = dtb[k * D_INNER + d];
    #pragma unroll
    for (int r = 0; r < DT_RANK; ++r) s += row[r] * wrow[r];
    delta[i] = (s > 20.f) ? s : log1pf(__expf(s));
}

// ---- selective scan: 16 lanes (one per state) per (b,k,d) channel ----------
__global__ void selective_scan(const float* __restrict__ delta,
                               const float* __restrict__ xsf,
                               const float* __restrict__ xdbl,
                               const float* __restrict__ A_logs,
                               const float* __restrict__ Ds,
                               float* __restrict__ outy)
{
    int t  = blockIdx.x * blockDim.x + threadIdx.x;
    int ch = t >> 4;                       // (b*KDIR + k)*D_INNER + d
    int n  = t & 15;                       // state index
    if (ch >= BATCH * KDIR * D_INNER) return;
    int d  = ch % D_INNER;
    int bk = ch / D_INNER;
    int k  = bk % KDIR;
    float An = -__expf(A_logs[((long long)(k * D_INNER + d)) * D_STATE + n]);
    float Dv = Ds[k * D_INNER + d];
    const float* del = delta + (long long)ch * LL;
    const float* u   = xsf   + (long long)ch * LL;
    const float* dbl = xdbl  + (long long)bk * LL * NPAD;
    float* out = outy + (long long)ch * LL;
    float hst = 0.f;
    for (int l = 0; l < LL; ++l) {
        // hide HBM latency on the serial dependency chain (global_prefetch_b8)
        __builtin_prefetch((const void*)(dbl + (l + 16) * NPAD), 0, 1);
        float dt = del[l];
        float ut = u[l];
        float Bt = dbl[l * NPAD + DT_RANK + n];
        float Ct = dbl[l * NPAD + DT_RANK + D_STATE + n];
        hst = hst * __expf(dt * An) + dt * ut * Bt;
        float yp = hst * Ct;
        yp += __shfl_xor(yp, 8, 32);       // reductions confined to 16-lane group
        yp += __shfl_xor(yp, 4, 32);
        yp += __shfl_xor(yp, 2, 32);
        yp += __shfl_xor(yp, 1, 32);
        if (n == 0) out[l] = yp + ut * Dv;
    }
}

// ---- fuse out_proj (96,192) with proj_w (96): v[d] = sum_c pw[c]*opw[c,d] --
__global__ void fuse_proj(const float* __restrict__ opw,
                          const float* __restrict__ pw,
                          float* __restrict__ v)
{
    int d = threadIdx.x;
    if (d >= D_INNER) return;
    float s = 0.f;
    for (int c = 0; c < DIMC; ++c) s += pw[c] * opw[(long long)c * D_INNER + d];
    v[d] = s;
}

__device__ __forceinline__ float block_reduce192(float val, float* sb)
{
    int tid = threadIdx.x;
    sb[tid] = val;
    __syncthreads();
    if (tid < 32) {
        float p = 0.f;
        #pragma unroll
        for (int i = 0; i < 6; ++i) p += sb[tid + i * 32];
        #pragma unroll
        for (int m = 16; m >= 1; m >>= 1) p += __shfl_xor(p, m, 32);
        if (tid == 0) sb[0] = p;
    }
    __syncthreads();
    float r = sb[0];
    __syncthreads();
    return r;
}

// ---- combine 4 directions, LayerNorm, SiLU gate, fused projection+sigmoid --
__global__ void combine_out(const float* __restrict__ outy,
                            const float* __restrict__ C0,
                            const float* __restrict__ gamma,
                            const float* __restrict__ beta,
                            const float* __restrict__ v,
                            const float* __restrict__ proj_b,
                            float* __restrict__ out)
{
    __shared__ float sb[D_INNER];
    int pix = blockIdx.x;                  // b*L + l
    int d   = threadIdx.x;
    int b = pix / LL, l = pix % LL;
    int h = l / WW, w = l % WW;
    int lT = w * HH + h;
    long long base = (long long)b * KDIR * D_INNER * LL;
    float y = outy[base + ((long long)(0 * D_INNER + d)) * LL + l]
            + outy[base + ((long long)(2 * D_INNER + d)) * LL + (LL - 1 - l)]
            + outy[base + ((long long)(1 * D_INNER + d)) * LL + lT]
            + outy[base + ((long long)(3 * D_INNER + d)) * LL + (LL - 1 - lT)];
    float mu = block_reduce192(y, sb) * (1.f / D_INNER);
    float m2 = block_reduce192(y * y, sb) * (1.f / D_INNER);
    float var = m2 - mu * mu;
    float yn = (y - mu) * rsqrtf(var + 1e-5f) * gamma[d] + beta[d];
    float z = C0[(long long)pix * (2 * D_INNER) + D_INNER + d];
    float g = yn * (z / (1.f + __expf(-z)));
    float dotv = block_reduce192(g * v[d], sb);
    if (d == 0) out[pix] = 1.f / (1.f + __expf(-(dotv + proj_b[0])));
}

// ---------------------------------------------------------------------------
extern "C" void kernel_launch(void* const* d_in, const int* in_sizes, int n_in,
                              void* d_out, int out_size, void* d_ws, size_t ws_size,
                              hipStream_t stream)
{
    const float* x         = (const float*)d_in[0];
    const float* in_proj_w = (const float*)d_in[1];
    const float* conv_w    = (const float*)d_in[2];
    const float* conv_b    = (const float*)d_in[3];
    const float* x_proj_w  = (const float*)d_in[4];
    const float* dt_w      = (const float*)d_in[5];
    const float* dt_b      = (const float*)d_in[6];
    const float* A_logs    = (const float*)d_in[7];
    const float* Ds        = (const float*)d_in[8];
    const float* gamma     = (const float*)d_in[9];
    const float* beta      = (const float*)d_in[10];
    const float* out_proj  = (const float*)d_in[11];
    const float* proj_w    = (const float*)d_in[12];
    const float* proj_b    = (const float*)d_in[13];
    float* out = (float*)d_out;

    // fragment-chunk counts (each chunk = one v16h = 32 bytes)
    const int nK0   = DIMC / 32;                      // 3
    const int tM0   = BATCH * LL / 16;                // 512
    const int tN0   = 2 * D_INNER / 16;               // 24
    const int nK1   = D_INNER / 32;                   // 6
    const int tM1   = LL / 16;                        // 256
    const int tN1   = NPAD / 16;                      // 3
    const long long chA0 = (long long)tM0 * nK0 * 32;
    const long long chB0 = (long long)tN0 * nK0 * 32;
    const long long chA1 = (long long)tM1 * nK1 * 32;          // per (b,k)
    const long long chB1 = (long long)tN1 * nK1 * 32;          // per k

    // workspace carve-up (all offsets 32B-aligned)
    char* ws = (char*)d_ws;
    size_t off = 0;
    float*    C0   = (float*)(ws + off);    off += (size_t)BATCH*LL*2*D_INNER*sizeof(float);
    v16h*     A0   = (v16h*)(ws + off);     off += (size_t)chA0 * sizeof(v16h);
    v16h*     B0   = (v16h*)(ws + off);     off += (size_t)chB0 * sizeof(v16h);
    float*    xc   = (float*)(ws + off);    off += (size_t)BATCH*D_INNER*LL*sizeof(float);
    float*    xsf  = (float*)(ws + off);    off += (size_t)BATCH*KDIR*D_INNER*LL*sizeof(float);
    v16h*     xsh  = (v16h*)(ws + off);     off += (size_t)BATCH*KDIR*chA1 * sizeof(v16h);
    v16h*     Bp   = (v16h*)(ws + off);     off += (size_t)KDIR*chB1 * sizeof(v16h);
    float*    xdbl = (float*)(ws + off);    off += (size_t)BATCH*KDIR*LL*NPAD*sizeof(float);
    float*    delt = (float*)(ws + off);    off += (size_t)BATCH*KDIR*D_INNER*LL*sizeof(float);
    float*    outy = (float*)(ws + off);    off += (size_t)BATCH*KDIR*D_INNER*LL*sizeof(float);
    float*    vfus = (float*)(ws + off);    off += (size_t)D_INNER*sizeof(float);
    (void)ws_size; (void)in_sizes; (void)n_in; (void)out_size;

    dim3 blk(256);

    // 1. in_proj GEMM (WMMA): (8192 x 96) * (96 x 384)
    pack_xA_frag<<<(int)((chA0 + 255) / 256), blk, 0, stream>>>(x, A0);
    pack_Bin_frag<<<(int)((chB0 + 255) / 256), blk, 0, stream>>>(in_proj_w, B0);
    {
        int waves = tM0 * tN0;
        wmma_gemm_frag<<<dim3((waves * 32 + 255) / 256, 1), blk, 0, stream>>>(
            A0, B0, C0, tM0, tN0, nK0, 2 * D_INNER, 0LL, 0LL, 0LL, 1);
    }

    // 2. depthwise conv + SiLU; 3. 4-direction sequences (scan u + A fragments)
    dwconv_silu<<<(BATCH*D_INNER*LL + 255)/256, blk, 0, stream>>>(C0, conv_w, conv_b, xc);
    build_xs<<<(BATCH*KDIR*D_INNER*LL + 255)/256, blk, 0, stream>>>(xc, xsf);
    pack_xs_frag<<<(int)((BATCH*KDIR*chA1 + 255) / 256), blk, 0, stream>>>(xc, xsh);

    // 4. x_dbl batched GEMM (WMMA): per (b,k): (4096 x 192) * (192 x 48pad)
    pack_xproj_frag<<<(int)((KDIR*chB1 + 255) / 256), blk, 0, stream>>>(x_proj_w, Bp);
    {
        int waves = tM1 * tN1;
        wmma_gemm_frag<<<dim3((waves * 32 + 255) / 256, BATCH * KDIR), blk, 0, stream>>>(
            xsh, Bp, xdbl, tM1, tN1, nK1, NPAD,
            chA1, chB1, (long long)LL * NPAD, KDIR);
    }

    // 5. delta; 6. selective scan; 7. fused tail
    compute_delta<<<(BATCH*KDIR*D_INNER*LL + 255)/256, blk, 0, stream>>>(xdbl, dt_w, dt_b, delt);
    selective_scan<<<(BATCH*KDIR*D_INNER*D_STATE + 255)/256, blk, 0, stream>>>(
        delt, xsf, xdbl, A_logs, Ds, outy);
    fuse_proj<<<1, D_INNER, 0, stream>>>(out_proj, proj_w, vfus);
    combine_out<<<BATCH*LL, D_INNER, 0, stream>>>(outy, C0, gamma, beta, vfus, proj_b, out);
}
